// SelfAttention_29566554865682
// MI455X (gfx1250) — matrix operand order repaired and checked by reading, not compiled, not run on hardware
//
#include <hip/hip_runtime.h>
#include <hip/hip_bf16.h>

typedef __attribute__((ext_vector_type(16))) __bf16   bf16x16;
typedef __attribute__((ext_vector_type(8)))  float    f32x8;
typedef __attribute__((ext_vector_type(8)))  unsigned u32x8;

static constexpr int    kB  = 16;
static constexpr int    kL  = 1024;
static constexpr int    kE  = 512;
static constexpr int    kBL = kB * kL;                      // 16384 flat rows
static constexpr size_t kMatElems = (size_t)kBL * kE;       // 8388608 elems per matrix
static constexpr size_t kWElems   = (size_t)kE * kE;        // 262144 elems per weight

// ---------- helpers ----------
__device__ __forceinline__ unsigned short f2bf_bits(float f) {
  unsigned u = __builtin_bit_cast(unsigned, f);
  u = (u + 0x7FFFu + ((u >> 16) & 1u)) >> 16;               // RNE truncate to bf16
  return (unsigned short)u;
}
__device__ __forceinline__ unsigned packbf(float lo, float hi) {
  return (unsigned)f2bf_bits(lo) | ((unsigned)f2bf_bits(hi) << 16);
}
__device__ __forceinline__ f32x8 wmma_bf16(u32x8 a, u32x8 b, f32x8 c) {
  return __builtin_amdgcn_wmma_f32_16x16x32_bf16(
      /*neg_a=*/false, __builtin_bit_cast(bf16x16, a),
      /*neg_b=*/false, __builtin_bit_cast(bf16x16, b),
      /*c_mod=*/(short)0, c, /*reuse_a=*/false, /*reuse_b=*/false);
}
__device__ __forceinline__ u32x8 load_frag32B(const unsigned short* p) {
  // 16 consecutive bf16 (32B, 16B-aligned) -> one operand fragment
  uint4 a = *(const uint4*)p;
  uint4 b = *(const uint4*)(p + 8);
  u32x8 r;
  r[0] = a.x; r[1] = a.y; r[2] = a.z; r[3] = a.w;
  r[4] = b.x; r[5] = b.y; r[6] = b.z; r[7] = b.w;
  return r;
}

// ============================================================================
// Kernel 0a: session f32 -> bf16, row-major (memory bound, once).
// ============================================================================
__global__ void __launch_bounds__(256)
cvt_bf16_kernel(const float* __restrict__ src, unsigned short* __restrict__ dst,
                int n4) {
  int i = blockIdx.x * 256 + (int)threadIdx.x;
  if (i >= n4) return;
  float4 v = ((const float4*)src)[i];
  uint2 o;
  o.x = packbf(v.x, v.y);
  o.y = packbf(v.z, v.w);
  ((uint2*)dst)[i] = o;
}

// ============================================================================
// Kernel 0b: weight f32 [k][n] -> bf16 transposed [n][k] (512KB once).
// ============================================================================
__global__ void __launch_bounds__(256)
cvt_wT_kernel(const float* __restrict__ w, unsigned short* __restrict__ wt) {
  int i = blockIdx.x * 256 + (int)threadIdx.x;     // 262144 threads
  int k = i >> 9, n = i & (kE - 1);
  wt[(size_t)n * kE + k] = f2bf_bits(w[i]);
}

// ============================================================================
// Kernel 1: Q/K/V = relu(session_bf16 @ W) using transposed bf16 weights.
// One 16x128 tile per wave (A fragment reused across 8 WMMAs).
// Q,K stored row-major; V stored TRANSPOSED [e][flat_row] with b128 stores.
// 3 * 1024 * 4 = 12288 waves.
// ============================================================================
__global__ void __launch_bounds__(128)
qkv_proj_kernel(const unsigned short* __restrict__ sess,
                const unsigned short* __restrict__ wt,   // wqT|wkT|wvT contiguous
                unsigned short* __restrict__ qkv) {
  int wave = blockIdx.x * 4 + ((int)threadIdx.x >> 5);
  int lane = (int)threadIdx.x & 31;
  if (wave >= 3 * (kBL / 16) * (kE / 128)) return;

  int mat  = wave >> 12;                // 4096 tiles per matrix
  int rem  = wave & 4095;
  int row0 = (rem >> 2) << 4;           // flat row tile (B*L dim)
  int c0   = (rem & 3) << 7;            // 128-wide output feature group
  const unsigned short* W = wt + (size_t)mat * kWElems;

  int m = lane & 15, h = lane >> 4;
  const unsigned short* arow = sess + (size_t)(row0 + m) * kE;

  f32x8 acc[8];
#pragma unroll
  for (int t = 0; t < 8; ++t) acc[t] = f32x8{0.f, 0.f, 0.f, 0.f, 0.f, 0.f, 0.f, 0.f};

  for (int k0 = 0; k0 < kE; k0 += 32) {
    // A fragment: row m, K = {k0+h*16+0..15}
    u32x8 au;
    uint4 a0 = *(const uint4*)(arow + k0 + h * 16);
    uint4 a1 = *(const uint4*)(arow + k0 + h * 16 + 8);
    au[0] = a0.x; au[1] = a0.y; au[2] = a0.z; au[3] = a0.w;
    au[4] = a1.x; au[5] = a1.y; au[6] = a1.z; au[7] = a1.w;
#pragma unroll
    for (int t = 0; t < 8; ++t) {
      // B fragment from W^T: column c0+t*16+m -> 16 consecutive K values
      u32x8 bu = load_frag32B(W + (size_t)(c0 + t * 16 + m) * kE + k0 + h * 16);
      acc[t] = wmma_bf16(au, bu, acc[t]);
    }
  }

  if (mat < 2) {
    // Q/K: ReLU + bf16 row-major store (C layout: VGPR j -> row h*8+j)
    unsigned short* dst = qkv + (size_t)mat * kMatElems;
#pragma unroll
    for (int t = 0; t < 8; ++t) {
#pragma unroll
      for (int j = 0; j < 8; ++j) {
        float v = acc[t][j] > 0.f ? acc[t][j] : 0.f;
        dst[(size_t)(row0 + h * 8 + j) * kE + (c0 + t * 16 + m)] = f2bf_bits(v);
      }
    }
  } else {
    // V: ReLU + bf16 TRANSPOSED store: lane holds 8 consecutive rows of col e
    unsigned short* vt = qkv + 2 * kMatElems;
#pragma unroll
    for (int t = 0; t < 8; ++t) {
      float r[8];
#pragma unroll
      for (int j = 0; j < 8; ++j) r[j] = acc[t][j] > 0.f ? acc[t][j] : 0.f;
      uint4 o;
      o.x = packbf(r[0], r[1]); o.y = packbf(r[2], r[3]);
      o.z = packbf(r[4], r[5]); o.w = packbf(r[6], r[7]);
      *(uint4*)(vt + (size_t)(c0 + t * 16 + m) * kBL + row0 + h * 8) = o;
    }
  }
}

// ============================================================================
// Kernel 2: flash attention. One wave: 16 query rows x 128 output columns.
// V is transposed [e][flat_row] so P@V B-fragments are contiguous loads.
// 16 batches * 64 q-tiles * 4 column slices = 4096 waves.
// ============================================================================
__global__ void __launch_bounds__(128)
flash_attn_kernel(const unsigned short* __restrict__ qkv,
                  const int* __restrict__ mask,
                  float* __restrict__ out) {
  int gw   = blockIdx.x * 4 + ((int)threadIdx.x >> 5);
  int lw   = (int)threadIdx.x >> 5;
  int lane = (int)threadIdx.x & 31;
  if (gw >= kB * (kL / 16) * (kE / 128)) return;

  int b  = gw >> 8;                 // 256 wave-jobs per batch
  int r  = gw & 255;
  int q0 = (r >> 2) << 4;           // query tile start (within batch)
  int e0 = (r & 3) << 7;            // output column slice start
  int m = lane & 15, h = lane >> 4;

  const unsigned short* Q  = qkv;
  const unsigned short* K  = qkv + kMatElems;
  const unsigned short* Vt = qkv + 2 * kMatElems;   // [e][flat_row]
  size_t rowb = (size_t)b * kL;

  __shared__ __align__(16) unsigned short pst[4][16][32];   // per-wave P staging

  f32x8 acc[8];
#pragma unroll
  for (int t = 0; t < 8; ++t) acc[t] = f32x8{0.f, 0.f, 0.f, 0.f, 0.f, 0.f, 0.f, 0.f};
  float runM[8], runS[8];
#pragma unroll
  for (int j = 0; j < 8; ++j) { runM[j] = -1e30f; runS[j] = 0.f; }

  const unsigned short* qrow = Q + (rowb + q0 + m) * (size_t)kE;
  const float scale = 0.044194173824159216f;   // 1/sqrt(512)

  for (int kb = 0; kb < kL; kb += 32) {
    // prefetch next key block's K rows (global_prefetch_b8)
    {
      int kn = (kb + 32) & (kL - 1);
      __builtin_prefetch(K + (rowb + kn + m) * (size_t)kE, 0, 3);
      __builtin_prefetch(K + (rowb + kn + 16 + m) * (size_t)kE, 0, 3);
    }
    // ---- S = Q @ K^T for 32 keys (two 16x16 C tiles) ----
    const unsigned short* kr0 = K + (rowb + kb + m) * (size_t)kE;
    const unsigned short* kr1 = K + (rowb + kb + 16 + m) * (size_t)kE;
    f32x8 s0 = {0.f, 0.f, 0.f, 0.f, 0.f, 0.f, 0.f, 0.f};
    f32x8 s1 = {0.f, 0.f, 0.f, 0.f, 0.f, 0.f, 0.f, 0.f};
    for (int k0 = 0; k0 < kE; k0 += 32) {
      u32x8 au;
      uint4 qa = *(const uint4*)(qrow + k0 + h * 16);
      uint4 qb = *(const uint4*)(qrow + k0 + h * 16 + 8);
      au[0] = qa.x; au[1] = qa.y; au[2] = qa.z; au[3] = qa.w;
      au[4] = qb.x; au[5] = qb.y; au[6] = qb.z; au[7] = qb.w;
      u32x8 b0u = load_frag32B(kr0 + k0 + h * 16);
      u32x8 b1u = load_frag32B(kr1 + k0 + h * 16);
      s0 = wmma_bf16(au, b0u, s0);
      s1 = wmma_bf16(au, b1u, s1);
    }

    // ---- online softmax over this 32-key block ----
    int mk0 = mask[rowb + kb + m];
    int mk1 = mask[rowb + kb + 16 + m];
    float p0[8], p1[8];
#pragma unroll
    for (int j = 0; j < 8; ++j) {
      float v0 = s0[j] * scale;
      float v1 = s1[j] * scale;
      float t = fmaxf(mk0 ? v0 : -1e30f, mk1 ? v1 : -1e30f);
#pragma unroll
      for (int off = 1; off < 16; off <<= 1)
        t = fmaxf(t, __shfl_xor(t, off, 32));      // row max within 16-lane group
      float nM   = fmaxf(runM[j], t);
      float corr = __expf(runM[j] - nM);
      float e0v  = mk0 ? __expf(v0 - nM) : 0.f;    // masked keys contribute 0
      float e1v  = mk1 ? __expf(v1 - nM) : 0.f;
      float ps = e0v + e1v;
#pragma unroll
      for (int off = 1; off < 16; off <<= 1)
        ps += __shfl_xor(ps, off, 32);             // row sum over 32 keys
      runS[j] = runS[j] * corr + ps;
      runM[j] = nM;
      p0[j] = e0v; p1[j] = e1v;
#pragma unroll
      for (int t2 = 0; t2 < 8; ++t2) acc[t2][j] *= corr;
    }

    // ---- stage P (C layout) -> LDS -> reload as A fragment ----
    asm volatile("" ::: "memory");
#pragma unroll
    for (int j = 0; j < 8; ++j) {
      pst[lw][h * 8 + j][m]      = f2bf_bits(p0[j]);
      pst[lw][h * 8 + j][16 + m] = f2bf_bits(p1[j]);
    }
    __builtin_amdgcn_wave_barrier();
    asm volatile("s_wait_dscnt 0" ::: "memory");
    u32x8 pu;
    {
      uint4 pa = *(const uint4*)&pst[lw][m][h * 16];
      uint4 pb = *(const uint4*)&pst[lw][m][h * 16 + 8];
      pu[0] = pa.x; pu[1] = pa.y; pu[2] = pa.z; pu[3] = pa.w;
      pu[4] = pb.x; pu[5] = pb.y; pu[6] = pb.z; pu[7] = pb.w;
    }
    __builtin_amdgcn_wave_barrier();

    // ---- O += P @ V over 8 output column tiles (contiguous V^T loads) ----
#pragma unroll
    for (int t2 = 0; t2 < 8; ++t2) {
      u32x8 vu = load_frag32B(Vt + (size_t)(e0 + t2 * 16 + m) * kBL +
                              rowb + kb + h * 16);
      acc[t2] = wmma_bf16(pu, vu, acc[t2]);
    }
  }

  // ---- epilogue: normalize, apply query-row mask, store f32 ----
#pragma unroll
  for (int j = 0; j < 8; ++j) {
    int qi  = q0 + h * 8 + j;
    int mq  = mask[rowb + qi];
    float inv = (mq != 0 && runS[j] > 0.f) ? 1.f / runS[j] : 0.f;
    float* orow = out + (rowb + qi) * (size_t)kE + e0 + m;
#pragma unroll
    for (int t2 = 0; t2 < 8; ++t2) orow[t2 * 16] = acc[t2][j] * inv;
  }
}

// ============================================================================
extern "C" void kernel_launch(void* const* d_in, const int* in_sizes, int n_in,
                              void* d_out, int out_size, void* d_ws, size_t ws_size,
                              hipStream_t stream) {
  (void)in_sizes; (void)n_in; (void)out_size; (void)ws_size;
  const float* session = (const float*)d_in[0];
  const int*   mask    = (const int*)d_in[1];
  const float* wq      = (const float*)d_in[2];
  const float* wk      = (const float*)d_in[3];
  const float* wv      = (const float*)d_in[4];
  float* out = (float*)d_out;

  // workspace layout (bf16 elems):
  //   [session_bf : 8388608][wqT|wkT|wvT : 3*262144][Q|K|V^T : 3*8388608]
  unsigned short* ws16  = (unsigned short*)d_ws;
  unsigned short* sessb = ws16;
  unsigned short* wtb   = ws16 + kMatElems;
  unsigned short* qkvb  = ws16 + kMatElems + 3 * kWElems;

  // Kernel 0: one-shot conversions
  cvt_bf16_kernel<<<(int)(kMatElems / 4 / 256), 256, 0, stream>>>(session, sessb,
                                                                  (int)(kMatElems / 4));
  cvt_wT_kernel<<<(int)(kWElems / 256), 256, 0, stream>>>(wq, wtb);
  cvt_wT_kernel<<<(int)(kWElems / 256), 256, 0, stream>>>(wk, wtb + kWElems);
  cvt_wT_kernel<<<(int)(kWElems / 256), 256, 0, stream>>>(wv, wtb + 2 * kWElems);
  // Kernel 1: 12288 tile-waves, 4 waves (128 threads) per block
  qkv_proj_kernel<<<3072, 128, 0, stream>>>(sessb, wtb, qkvb);
  // Kernel 2: 4096 attention waves
  flash_attn_kernel<<<1024, 128, 0, stream>>>(qkvb, mask, out);
}
